// ZayaBlock_61830349193728
// MI455X (gfx1250) — matrix operand order repaired and checked by reading, not compiled, not run on hardware
//
#include <hip/hip_runtime.h>
#include <math.h>

#define S_TOK 4096
#define HID   2048
#define DR    256
#define NE    8
#define IS    1408
#define EPSR  1e-5f

typedef __attribute__((ext_vector_type(16))) __bf16 v16bf;
typedef __attribute__((ext_vector_type(8)))  float  v8f;
typedef __attribute__((ext_vector_type(4)))  float  f32x4;
typedef __attribute__((ext_vector_type(4)))  __bf16 bf16x4;

union FragAB { v16bf v; unsigned int u[8]; };
union FragC  { v8f   v; float        f[8]; };

#define ASTRIDE 40   // padded LDS row stride (bf16 elems) for 32-wide K tiles
#define BSTRIDE 40

// convert 4 consecutive f32 -> 4 packed bf16 (2x v_cvt_pk_bf16_f32, no repack)
__device__ inline uint2 cvt4(float4 f) {
  f32x4 v; v.x = f.x; v.y = f.y; v.z = f.z; v.w = f.w;
  bf16x4 h = __builtin_convertvector(v, bf16x4);
  union { bf16x4 h; uint2 u; } c; c.h = h;
  return c.u;
}

// A fragment: 16x32 bf16, lane(0..15)=row M, VGPR j holds K pair 16*(lane/16)+2j
__device__ inline v16bf frag_a(const __bf16* tile, int row0, int lane) {
  const __bf16* row = tile + (size_t)(row0 + (lane & 15)) * ASTRIDE;
  const int kg = (lane >> 4) * 16;
  FragAB f;
#pragma unroll
  for (int j = 0; j < 8; ++j) {
    int kb = 2 * j + kg;
    f.u[j] = *(const unsigned int*)(row + kb);
  }
  return f.v;
}

// B fragment: 32x16 bf16 (stored as W[n][k] rows), lane(0..15)=col N
__device__ inline v16bf frag_b(const __bf16* tile, int col0, int lane) {
  const __bf16* row = tile + (size_t)(col0 + (lane & 15)) * BSTRIDE;
  const int kh = (lane >> 4) * 16;
  FragAB f;
#pragma unroll
  for (int j = 0; j < 8; ++j)
    f.u[j] = *(const unsigned int*)(row + kh + 2 * j);
  return f.v;
}

__device__ inline v8f wmma_bf16(v16bf a, v16bf b, v8f c) {
  return __builtin_amdgcn_wmma_f32_16x16x32_bf16(
      false, a, false, b, (short)0, c, false, false);
}

// register-staged tiles (software pipeline)
struct St16 { float4 a, b, c, d; };
struct St8  { float4 a, b; };

__device__ inline St16 ld16f(const float* p) {
  St16 s;
  s.a = ((const float4*)p)[0]; s.b = ((const float4*)p)[1];
  s.c = ((const float4*)p)[2]; s.d = ((const float4*)p)[3];
  return s;
}
__device__ inline St8 ld8f(const float* p) {
  St8 s;
  s.a = ((const float4*)p)[0]; s.b = ((const float4*)p)[1];
  return s;
}
__device__ inline void st16b(const St16& s, __bf16* dst) {
  uint2 c0 = cvt4(s.a), c1 = cvt4(s.b), c2 = cvt4(s.c), c3 = cvt4(s.d);
  uint4 q0, q1;
  q0.x = c0.x; q0.y = c0.y; q0.z = c1.x; q0.w = c1.y;
  q1.x = c2.x; q1.y = c2.y; q1.z = c3.x; q1.w = c3.y;
  ((uint4*)dst)[0] = q0;
  ((uint4*)dst)[1] = q1;
}
__device__ inline void st8b(const St8& s, __bf16* dst) {
  uint2 c0 = cvt4(s.a), c1 = cvt4(s.b);
  uint4 q;
  q.x = c0.x; q.y = c0.y; q.z = c1.x; q.w = c1.y;
  *(uint4*)dst = q;
}

// CDNA5 async global -> LDS copy (16B per lane, tracked by ASYNCcnt)
__device__ inline void async_ld_b128(unsigned lds_off, const void* gaddr) {
  asm volatile("global_load_async_to_lds_b128 %0, %1, off"
               :: "v"(lds_off), "v"((unsigned long long)(uintptr_t)gaddr)
               : "memory");
}
__device__ inline void wait_async0() {
  asm volatile("s_wait_asynccnt 0x0" ::: "memory");
}

// ---------------------------------------------------------------------------
// Router down-proj: C[M,N] = A[M,K] * W[N,K]^T + bias
// Block 128x128, 8 waves (2 M x 4 N), wave tile 64x32, K step 32, pipelined.
// ---------------------------------------------------------------------------
__global__ __launch_bounds__(256)
void gemm_bias(const float* __restrict__ A, int lda,
               const float* __restrict__ W, int ldw,
               const float* __restrict__ bias,
               float* __restrict__ C, int ldc, int K) {
  __shared__ __bf16 As[128 * ASTRIDE];
  __shared__ __bf16 Bs[128 * BSTRIDE];
  const int tid  = threadIdx.x;
  const int lane = tid & 31;
  const int wave = tid >> 5;
  const int wm   = wave >> 2;
  const int wn   = wave & 3;
  const int m0   = blockIdx.x * 128;
  const int n0   = blockIdx.y * 128;
  const int srow = tid >> 1;
  const int scol = (tid & 1) * 16;

  v8f acc[4][2] = {};

  const float* apos = A + (size_t)(m0 + srow) * lda + scol;
  const float* wpos = W + (size_t)(n0 + srow) * ldw + scol;
  __bf16* adst = &As[srow * ASTRIDE + scol];
  __bf16* bdst = &Bs[srow * BSTRIDE + scol];

  auto mac_tile = [&]() {
    v16bf a0 = frag_a(As, wm * 64,      lane);
    v16bf a1 = frag_a(As, wm * 64 + 16, lane);
    v16bf a2 = frag_a(As, wm * 64 + 32, lane);
    v16bf a3 = frag_a(As, wm * 64 + 48, lane);
    v16bf b0 = frag_b(Bs, wn * 32,      lane);
    v16bf b1 = frag_b(Bs, wn * 32 + 16, lane);
    acc[0][0] = wmma_bf16(a0, b0, acc[0][0]);
    acc[0][1] = wmma_bf16(a0, b1, acc[0][1]);
    acc[1][0] = wmma_bf16(a1, b0, acc[1][0]);
    acc[1][1] = wmma_bf16(a1, b1, acc[1][1]);
    acc[2][0] = wmma_bf16(a2, b0, acc[2][0]);
    acc[2][1] = wmma_bf16(a2, b1, acc[2][1]);
    acc[3][0] = wmma_bf16(a3, b0, acc[3][0]);
    acc[3][1] = wmma_bf16(a3, b1, acc[3][1]);
  };

  St16 ra = ld16f(apos);
  St16 rb = ld16f(wpos);
  st16b(ra, adst);
  st16b(rb, bdst);
  __syncthreads();

  for (int k0 = 32; k0 < K; k0 += 32) {
    ra = ld16f(apos + k0);           // next tile, overlaps compute
    rb = ld16f(wpos + k0);
    __builtin_prefetch(wpos + k0 + 32, 0, 0);
    mac_tile();
    __syncthreads();
    st16b(ra, adst);
    st16b(rb, bdst);
    __syncthreads();
  }
  mac_tile();

  const int mlo = (lane >> 4) * 8;
  const int nn  = lane & 15;
#pragma unroll
  for (int mi = 0; mi < 4; ++mi) {
#pragma unroll
    for (int ni = 0; ni < 2; ++ni) {
      FragC c; c.v = acc[mi][ni];
#pragma unroll
      for (int r = 0; r < 8; ++r) {
        int row = m0 + wm * 64 + mi * 16 + mlo + r;
        int col = n0 + wn * 32 + ni * 16 + nn;
        C[(size_t)row * ldc + col] = c.f[r] + bias[col];
      }
    }
  }
}

// ---------------------------------------------------------------------------
// Router tail: RMSNorm -> GELU MLP -> logits -> softmax -> top2 ->
// append token to its 2 experts' gather lists (atomic counters).
// ---------------------------------------------------------------------------
__global__ __launch_bounds__(256)
void router_mlp(const float* __restrict__ hs, const float* __restrict__ rms_w,
                const float* __restrict__ Wm1, const float* __restrict__ bm1,
                const float* __restrict__ Wm2, const float* __restrict__ bm2,
                const float* __restrict__ Wr,
                int* __restrict__ counts, int* __restrict__ tok_list,
                float* __restrict__ gate_w) {
  const int s = blockIdx.x;
  const int t = threadIdx.x;
  __shared__ float xa[DR];
  __shared__ float xb[DR];
  __shared__ float wsum[8];
  __shared__ float logits[NE];
  __shared__ float scale_s;

  float v = hs[(size_t)s * DR + t];
  float sq = v * v;
#pragma unroll
  for (int o = 16; o > 0; o >>= 1) sq += __shfl_xor(sq, o, 32);
  if ((t & 31) == 0) wsum[t >> 5] = sq;
  __syncthreads();
  if (t == 0) {
    float tot = 0.f;
    for (int i = 0; i < 8; ++i) tot += wsum[i];
    scale_s = rsqrtf(tot / (float)DR + EPSR);
  }
  __syncthreads();
  xa[t] = v * scale_s * rms_w[t];
  __syncthreads();
  {
    const float* wr = Wm1 + (size_t)t * DR;
    float acc = bm1[t];
    for (int j = 0; j < DR; ++j) acc += xa[j] * wr[j];
    xb[t] = 0.5f * acc * (1.f + erff(acc * 0.70710678118654752f));
  }
  __syncthreads();
  {
    const float* wr = Wm2 + (size_t)t * DR;
    float acc = bm2[t];
    for (int j = 0; j < DR; ++j) acc += xb[j] * wr[j];
    xa[t] = 0.5f * acc * (1.f + erff(acc * 0.70710678118654752f));
  }
  __syncthreads();
  if (t < NE) {
    const float* wr = Wr + (size_t)t * DR;
    float acc = 0.f;
    for (int j = 0; j < DR; ++j) acc += xa[j] * wr[j];
    logits[t] = acc;
  }
  __syncthreads();
  if (t == 0) {
    float mx = logits[0];
    for (int i = 1; i < NE; ++i) mx = fmaxf(mx, logits[i]);
    float p[NE]; float sum = 0.f;
    for (int i = 0; i < NE; ++i) { p[i] = __expf(logits[i] - mx); sum += p[i]; }
    float inv = __builtin_amdgcn_rcpf(sum);
    int i1 = 0; float b1 = -1.f;
    for (int i = 0; i < NE; ++i) {
      p[i] *= inv;
      if (p[i] > b1) { b1 = p[i]; i1 = i; }
    }
    int i2 = 0; float b2 = -1.f;
    for (int i = 0; i < NE; ++i)
      if (i != i1 && p[i] > b2) { b2 = p[i]; i2 = i; }
    int p1 = atomicAdd(&counts[i1], 1);
    tok_list[(size_t)i1 * S_TOK + p1] = s;
    gate_w[(size_t)i1 * S_TOK + p1] = b1;
    int p2 = atomicAdd(&counts[i2], 1);
    tok_list[(size_t)i2 * S_TOK + p2] = s;
    gate_w[(size_t)i2 * S_TOK + p2] = b2;
  }
}

// ---------------------------------------------------------------------------
// Gathered fused gate/up for one expert:
//   act[pos, i] = silu(x[tok]@Wg^T) * (x[tok]@Wu^T) * gate_w[pos]  (bf16)
// Block 128(M tokens) x 64(N within I); 8 waves (4 M x 2 N), dual 32x32 tiles.
// ---------------------------------------------------------------------------
__global__ __launch_bounds__(256)
void moe_gate_up(const float* __restrict__ X,
                 const float* __restrict__ w13e,      // (2I, H)
                 const int* __restrict__ counts, int e,
                 const int* __restrict__ tok_list_e,  // (S)
                 const float* __restrict__ gate_w_e,  // (S)
                 __bf16* __restrict__ act) {          // (S, I) compact rows
  const int cnt  = counts[e];
  const int base = blockIdx.x * 128;
  if (base >= cnt) return;

  __shared__ int    ttok[128];
  __shared__ float  tgw[128];
  __shared__ __bf16 As[128 * ASTRIDE];
  __shared__ __bf16 Bg[64 * BSTRIDE];
  __shared__ __bf16 Bu[64 * BSTRIDE];

  const int tid  = threadIdx.x;
  const int lane = tid & 31;
  const int wave = tid >> 5;
  const int wm   = wave >> 1;        // 0..3
  const int wn   = wave & 1;         // 0..1
  const int n0   = blockIdx.y * 64;  // within I

  if (tid < 128) {
    int p  = base + tid;
    int pc = (p < cnt) ? p : (cnt - 1);
    ttok[tid] = tok_list_e[pc];
    tgw[tid]  = (p < cnt) ? gate_w_e[p] : 0.f;
  }
  __syncthreads();

  const int arow = tid >> 1;          // 0..127
  const int acol = (tid & 1) * 16;    // 0,16
  const int brow = tid >> 2;          // 0..63
  const int bcol = (tid & 3) * 8;     // 0,8,16,24
  const int atok = ttok[arow];

  v8f accg[2][2] = {};
  v8f accu[2][2] = {};

  const float* apos = X + (size_t)atok * HID + acol;
  const float* gpos = w13e + (size_t)(n0 + brow) * HID + bcol;
  const float* upos = gpos + (size_t)IS * HID;
  __bf16* adst = &As[arow * ASTRIDE + acol];
  __bf16* gdst = &Bg[brow * BSTRIDE + bcol];
  __bf16* udst = &Bu[brow * BSTRIDE + bcol];

  auto mac_tile = [&]() {
    v16bf a0  = frag_a(As, wm * 32,      lane);
    v16bf a1  = frag_a(As, wm * 32 + 16, lane);
    v16bf bg0 = frag_b(Bg, wn * 32,      lane);
    v16bf bg1 = frag_b(Bg, wn * 32 + 16, lane);
    v16bf bu0 = frag_b(Bu, wn * 32,      lane);
    v16bf bu1 = frag_b(Bu, wn * 32 + 16, lane);
    accg[0][0] = wmma_bf16(a0, bg0, accg[0][0]);
    accg[0][1] = wmma_bf16(a0, bg1, accg[0][1]);
    accg[1][0] = wmma_bf16(a1, bg0, accg[1][0]);
    accg[1][1] = wmma_bf16(a1, bg1, accg[1][1]);
    accu[0][0] = wmma_bf16(a0, bu0, accu[0][0]);
    accu[0][1] = wmma_bf16(a0, bu1, accu[0][1]);
    accu[1][0] = wmma_bf16(a1, bu0, accu[1][0]);
    accu[1][1] = wmma_bf16(a1, bu1, accu[1][1]);
  };

  St16 ra = ld16f(apos);
  St8  rg = ld8f(gpos);
  St8  ru = ld8f(upos);
  st16b(ra, adst); st8b(rg, gdst); st8b(ru, udst);
  __syncthreads();

  for (int k0 = 32; k0 < HID; k0 += 32) {
    ra = ld16f(apos + k0);
    rg = ld8f(gpos + k0);
    ru = ld8f(upos + k0);
    __builtin_prefetch(gpos + k0 + 32, 0, 0);
    __builtin_prefetch(upos + k0 + 32, 0, 0);
    mac_tile();
    __syncthreads();
    st16b(ra, adst); st8b(rg, gdst); st8b(ru, udst);
    __syncthreads();
  }
  mac_tile();

  const int mlo = (lane >> 4) * 8;
  const int nn  = lane & 15;
#pragma unroll
  for (int mi = 0; mi < 2; ++mi) {
#pragma unroll
    for (int ni = 0; ni < 2; ++ni) {
      FragC cg; cg.v = accg[mi][ni];
      FragC cu; cu.v = accu[mi][ni];
#pragma unroll
      for (int r = 0; r < 8; ++r) {
        int lrow = wm * 32 + mi * 16 + mlo + r;
        int col  = n0 + wn * 32 + ni * 16 + nn;
        float g = cg.f[r], u = cu.f[r];
        float sig = __builtin_amdgcn_rcpf(1.f + __expf(-g));
        float a = g * sig * u * tgw[lrow];      // silu(g)*u*prob
        act[(size_t)(base + lrow) * IS + col] = (__bf16)a;
      }
    }
  }
}

// ---------------------------------------------------------------------------
// Gathered down-proj for one expert: y[tok, h] += act[pos, :] @ w2e[h, :]^T
// Block 128(M pos) x 128(N=H), 8 waves (2 M x 4 N), wave tile 64x32.
// A operand (bf16, no conversion) streamed with ASYNC global->LDS copies
// into a double buffer; weights pipelined through registers.
// ---------------------------------------------------------------------------
__global__ __launch_bounds__(256)
void moe_down(const __bf16* __restrict__ act,       // (S, I) compact rows
              const float* __restrict__ w2e,        // (H, I)
              const int* __restrict__ counts, int e,
              const int* __restrict__ tok_list_e,
              float* __restrict__ y) {              // (S, H)
  const int cnt  = counts[e];
  const int base = blockIdx.x * 128;
  if (base >= cnt) return;

  __shared__ int    ttok[128];
  __shared__ __bf16 As[2][128 * ASTRIDE];
  __shared__ __bf16 Bs[128 * BSTRIDE];

  const int tid  = threadIdx.x;
  const int lane = tid & 31;
  const int wave = tid >> 5;
  const int wm   = wave >> 2;        // 0..1
  const int wn   = wave & 3;         // 0..3
  const int n0   = blockIdx.y * 128;

  if (tid < 128) {
    int p = base + tid;
    ttok[tid] = tok_list_e[(p < cnt) ? p : (cnt - 1)];
  }
  __syncthreads();

  const int srow = tid >> 1;
  const int scol = (tid & 1) * 16;

  v8f acc[4][2] = {};

  const __bf16* apos = act + (size_t)(base + srow) * IS + scol;
  const float*  wpos = w2e + (size_t)(n0 + srow) * IS + scol;
  __bf16* bdst = &Bs[srow * BSTRIDE + scol];
  const unsigned lds_a0 =
      (unsigned)(unsigned long long)(uintptr_t)&As[0][srow * ASTRIDE + scol];
  const unsigned lds_a1 =
      (unsigned)(unsigned long long)(uintptr_t)&As[1][srow * ASTRIDE + scol];

  auto mac_tile = [&](const __bf16* Ab) {
    v16bf a0 = frag_a(Ab, wm * 64,      lane);
    v16bf a1 = frag_a(Ab, wm * 64 + 16, lane);
    v16bf a2 = frag_a(Ab, wm * 64 + 32, lane);
    v16bf a3 = frag_a(Ab, wm * 64 + 48, lane);
    v16bf b0 = frag_b(Bs, wn * 32,      lane);
    v16bf b1 = frag_b(Bs, wn * 32 + 16, lane);
    acc[0][0] = wmma_bf16(a0, b0, acc[0][0]);
    acc[0][1] = wmma_bf16(a0, b1, acc[0][1]);
    acc[1][0] = wmma_bf16(a1, b0, acc[1][0]);
    acc[1][1] = wmma_bf16(a1, b1, acc[1][1]);
    acc[2][0] = wmma_bf16(a2, b0, acc[2][0]);
    acc[2][1] = wmma_bf16(a2, b1, acc[2][1]);
    acc[3][0] = wmma_bf16(a3, b0, acc[3][0]);
    acc[3][1] = wmma_bf16(a3, b1, acc[3][1]);
  };

  // prologue: async A tile 0, register-staged B tile 0
  async_ld_b128(lds_a0,      apos);
  async_ld_b128(lds_a0 + 16, apos + 8);
  St16 rb = ld16f(wpos);
  st16b(rb, bdst);
  wait_async0();
  __syncthreads();

  int buf = 0;
  for (int k0 = 32; k0 < IS; k0 += 32) {
    unsigned dsta = buf ? lds_a0 : lds_a1;       // next A buffer
    async_ld_b128(dsta,      apos + k0);
    async_ld_b128(dsta + 16, apos + k0 + 8);
    rb = ld16f(wpos + k0);
    __builtin_prefetch(wpos + k0 + 32, 0, 0);
    mac_tile(As[buf]);
    __syncthreads();          // all waves done reading Bs / As[buf]
    st16b(rb, bdst);
    wait_async0();            // next A tile landed in LDS
    __syncthreads();
    buf ^= 1;
  }
  mac_tile(As[buf]);

  const int mlo = (lane >> 4) * 8;
  const int nn  = lane & 15;
#pragma unroll
  for (int mi = 0; mi < 4; ++mi) {
#pragma unroll
    for (int ni = 0; ni < 2; ++ni) {
      FragC c; c.v = acc[mi][ni];
#pragma unroll
      for (int r = 0; r < 8; ++r) {
        int lrow = wm * 64 + mi * 16 + mlo + r;
        if (base + lrow < cnt) {
          int tok = ttok[lrow];
          int col = n0 + wn * 32 + ni * 16 + nn;
          y[(size_t)tok * HID + col] += c.f[r];   // scatter-accumulate
        }
      }
    }
  }
}

// ---------------------------------------------------------------------------
extern "C" void kernel_launch(void* const* d_in, const int* in_sizes, int n_in,
                              void* d_out, int out_size, void* d_ws, size_t ws_size,
                              hipStream_t stream) {
  (void)in_sizes; (void)n_in; (void)out_size; (void)ws_size;
  const float* x   = (const float*)d_in[0];
  const float* Wd  = (const float*)d_in[1];
  const float* bd  = (const float*)d_in[2];
  const float* rms = (const float*)d_in[3];
  const float* Wm1 = (const float*)d_in[4];
  const float* bm1 = (const float*)d_in[5];
  const float* Wm2 = (const float*)d_in[6];
  const float* bm2 = (const float*)d_in[7];
  const float* Wr  = (const float*)d_in[8];
  const float* w13 = (const float*)d_in[9];
  const float* w2  = (const float*)d_in[10];
  // d_in[11] = topk (fixed 2 in this config)

  float* y           = (float*)d_out;                    // (S, H)
  float* router_next = y + (size_t)S_TOK * HID;          // (S, D)

  // workspace layout
  int*   counts   = (int*)d_ws;                                  // NE (pad to 16)
  int*   tok_list = (int*)d_ws + 16;                             // NE*S
  float* gate_w   = (float*)((int*)d_ws + 16 + NE * S_TOK);      // NE*S
  __bf16* act = (__bf16*)((char*)d_ws +
                          (16 + 2 * NE * S_TOK) * sizeof(int));  // S*I bf16

  hipMemsetAsync(counts, 0, 16 * sizeof(int), stream);
  hipMemsetAsync(y, 0, (size_t)S_TOK * HID * sizeof(float), stream);

  dim3 blk(256);
  gemm_bias<<<dim3(S_TOK / 128, DR / 128), blk, 0, stream>>>(
      x, HID, Wd, HID, bd, router_next, DR, HID);

  router_mlp<<<dim3(S_TOK), blk, 0, stream>>>(
      router_next, rms, Wm1, bm1, Wm2, bm2, Wr, counts, tok_list, gate_w);

  for (int e = 0; e < NE; ++e) {
    const float* w13e = w13 + (size_t)e * 2 * IS * HID;
    const float* w2e  = w2  + (size_t)e * HID * IS;
    const int*   tle  = tok_list + (size_t)e * S_TOK;
    const float* gwe  = gate_w + (size_t)e * S_TOK;
    moe_gate_up<<<dim3(S_TOK / 128, IS / 64), blk, 0, stream>>>(
        x, w13e, counts, e, tle, gwe, act);
    moe_down<<<dim3(S_TOK / 128, HID / 128), blk, 0, stream>>>(
        act, w2e, counts, e, tle, y);
  }
}
